// QuantumNatEnhanced_65481071403504
// MI455X (gfx1250) — matrix operand order repaired and checked by reading, not compiled, not run on hardware
//
#include <hip/hip_runtime.h>
#include <hip/hip_bf16.h>

typedef __attribute__((ext_vector_type(16))) _Float16 v16h;
typedef __attribute__((ext_vector_type(8)))  float    v8f;

__device__ __forceinline__ float2 cmul(float2 a, float2 b){
  return make_float2(a.x*b.x - a.y*b.y, a.x*b.y + a.y*b.x);
}
__device__ __forceinline__ float2 cadd(float2 a, float2 b){
  return make_float2(a.x + b.x, a.y + b.y);
}
__device__ __forceinline__ v16h pack16(uint4 lo, uint4 hi){
  union { uint4 u[2]; v16h h; } t;
  t.u[0] = lo; t.u[1] = hi;
  return t.h;
}

// ---------------- stage 1: direct f32 conv 1->16 (K=9, too small for WMMA) ----------------
__global__ __launch_bounds__(256) void conv1_kernel(
    const float* __restrict__ x, const float* __restrict__ w,
    const float* __restrict__ bias, float* __restrict__ y,
    float* __restrict__ sums)
{
  __shared__ float ls[32];
  int tid = threadIdx.x;
  if (tid < 32) ls[tid] = 0.f;
  __syncthreads();
  int idx = blockIdx.x*256 + tid;            // (b,c,oh,ow) over 1024*16*28*28
  int ow = idx % 28;
  int oh = (idx/28) % 28;
  int c  = (idx/784) % 16;
  int b  = idx/12544;
  float acc = bias[c];
  #pragma unroll
  for (int kh = 0; kh < 3; ++kh){
    int ih = oh + kh - 1;
    #pragma unroll
    for (int kw = 0; kw < 3; ++kw){
      int iw = ow + kw - 1;
      if ((unsigned)ih < 28u && (unsigned)iw < 28u)
        acc += x[b*784 + ih*28 + iw] * w[c*9 + kh*3 + kw];
    }
  }
  y[idx] = acc;
  atomicAdd(&ls[c], acc);
  atomicAdd(&ls[16 + c], acc*acc);
  __syncthreads();
  if (tid < 32) atomicAdd(&sums[tid], ls[tid]);
}

// ---------------- implicit-GEMM conv via v_wmma_f32_16x16x32_f16 ----------------
// GEMM: M = B*H*W, N = Cout, K = Cin*9. One wave = 16x16 tile. Weights staged in LDS.
template<int Cin, int H, int W, int Cout>
__global__ __launch_bounds__(256) void conv_wmma_kernel(
    const _Float16* __restrict__ act,   // (B,Cin,H,W) f16
    const _Float16* __restrict__ wgt,   // (Cout, Cin*9) f16
    const float* __restrict__ bias,     // (Cout)
    float* __restrict__ y,              // (B,Cout,H,W) f32 (pre-BN)
    float* __restrict__ sums)           // [0,Cout)=sum  [Cout,2Cout)=sumsq
{
  constexpr int K  = Cin*9;
  constexpr int NT = Cout/16;
  __shared__ _Float16 wsm[Cout*K];
  __shared__ float ls[2*Cout];
  int tid = threadIdx.x;
  {
    constexpr int NV = (Cout*K)/8;               // 16B vectors
    const uint4* gsrc = (const uint4*)wgt;
    uint4* ldst = (uint4*)wsm;
    for (int i = tid; i < NV; i += 256) ldst[i] = gsrc[i];
  }
  if (tid < 2*Cout) ls[tid] = 0.f;
  __syncthreads();

  int wave   = blockIdx.x*8 + (tid >> 5);
  int lane   = tid & 31;
  int laneHi = lane >> 4;
  int lm     = lane & 15;
  int mt = wave / NT;                           // NT is power of two -> shift
  int nt = wave & (NT - 1);

  int m  = mt*16 + lm;                          // A row held by this lane
  int ow = m % W; int oh = (m/W) % H; int b = m/(W*H);
  int n  = nt*16 + lm;                          // output channel of this lane (B/D col)
  int kbaseA = laneHi ? 8  : 0;
  int kbaseB = laneHi ? 8  : 0;

  v8f c = {};
  #pragma unroll
  for (int k0 = 0; k0 < K; k0 += 32){
    v16h af;
    #pragma unroll
    for (int e = 0; e < 16; ++e){
      int ka = k0 + kbaseA + (e < 8 ? e : e + 8);   // A-fragment K pattern (ISA 7.12.2)
      _Float16 av = (_Float16)0.f;
      if (ka < K){
        int ci = ka/9, r = ka - ci*9, kh = r/3, kw = r - kh*3;
        int ih = oh + kh - 1, iw = ow + kw - 1;
        if ((unsigned)ih < (unsigned)H && (unsigned)iw < (unsigned)W)
          av = act[((b*Cin + ci)*H + ih)*W + iw];
      }
      af[e] = av;
    }
    // B fragment: N = lane%16, K = k0 + 8*laneHi + (e<8 ? e : e+8) -> two 16B runs in LDS
    const uint4* bp = (const uint4*)&wsm[n*K + k0 + kbaseB];
    v16h bf = pack16(bp[0], bp[2]);
    c = __builtin_amdgcn_wmma_f32_16x16x32_f16(false, af, false, bf, (short)0, c, false, false);
  }

  float bval = bias[n];
  float lsum = 0.f, lsq = 0.f;
  #pragma unroll
  for (int r = 0; r < 8; ++r){
    int mr = mt*16 + r + (laneHi ? 8 : 0);
    float v = c[r] + bval;
    int owr = mr % W; int ohr = (mr/W) % H; int br = mr/(W*H);
    y[((br*Cout + n)*H + ohr)*W + owr] = v;
    lsum += v; lsq += v*v;
  }
  atomicAdd(&ls[n], lsum);          // each lane owns a single channel n -> 2 LDS atomics
  atomicAdd(&ls[Cout + n], lsq);
  __syncthreads();
  if (tid < 2*Cout) atomicAdd(&sums[tid], ls[tid]);
}

// ---------------- fold batch stats into per-channel scale/shift ----------------
__global__ void bn_finalize_kernel(const float* __restrict__ sums,
    const float* __restrict__ g, const float* __restrict__ beta,
    float* __restrict__ ss, int C, float invN)
{
  int c = threadIdx.x;
  if (c < C){
    float mean = sums[c]*invN;
    float var  = sums[C + c]*invN - mean*mean;
    float sc   = g[c]*rsqrtf(var + 1e-5f);
    ss[c]      = sc;
    ss[C + c]  = beta[c] - mean*sc;
  }
}

// ---------------- fused BN + ReLU + 2x2 maxpool, downcast to f16 ----------------
template<int C, int H, int W, int OH, int OW>
__global__ __launch_bounds__(256) void bn_relu_pool_kernel(
    const float* __restrict__ y, const float* __restrict__ ss,
    _Float16* __restrict__ out)
{
  int idx = blockIdx.x*256 + threadIdx.x;    // (b,c,oh,ow)
  int ow = idx % OW;
  int oh = (idx/OW) % OH;
  int c  = (idx/(OW*OH)) % C;
  int b  = idx/(OW*OH*C);
  float sc = ss[c], sh = ss[C + c];
  const float* p0 = y + ((size_t)(b*C + c)*H + 2*oh)*W + 2*ow;
  float v0 = fmaxf(p0[0]    *sc + sh, 0.f);
  float v1 = fmaxf(p0[1]    *sc + sh, 0.f);
  float v2 = fmaxf(p0[W]    *sc + sh, 0.f);
  float v3 = fmaxf(p0[W + 1]*sc + sh, 0.f);
  out[idx] = (_Float16)fmaxf(fmaxf(v0, v1), fmaxf(v2, v3));
}

// ---------------- linear 576->24 via WMMA (N padded 24->32) ----------------
__global__ __launch_bounds__(256) void linear_wmma_kernel(
    const _Float16* __restrict__ A,   // (1024, 576) f16  (flattened pool3 output)
    const _Float16* __restrict__ Wh,  // (24, 576) f16
    const float* __restrict__ bias,   // (24)
    float* __restrict__ P)            // (1024, 24) f32
{
  constexpr int K = 576, N = 24;
  int tid    = threadIdx.x;
  int wave   = blockIdx.x*8 + (tid >> 5);
  int lane   = tid & 31;
  int laneHi = lane >> 4;
  int lm     = lane & 15;
  int mt = wave >> 1;
  int nt = wave & 1;
  int m  = mt*16 + lm;
  int n  = nt*16 + lm;
  int kbaseA = laneHi ? 8  : 0;
  int kbaseB = laneHi ? 8  : 0;
  uint4 zz = make_uint4(0u, 0u, 0u, 0u);
  v8f c = {};
  #pragma unroll
  for (int k0 = 0; k0 < K; k0 += 32){
    // A fragment: two contiguous 16B runs at k0+kbaseA and k0+kbaseA+16
    const uint4* ap = (const uint4*)&A[m*K + k0 + kbaseA];
    v16h af = pack16(ap[0], ap[2]);
    // B fragment: two contiguous 16B runs at k0+kbaseB and k0+kbaseB+16
    v16h bf;
    if (n < N){
      const uint4* bp = (const uint4*)&Wh[n*K + k0 + kbaseB];
      bf = pack16(bp[0], bp[2]);
    } else {
      bf = pack16(zz, zz);
    }
    c = __builtin_amdgcn_wmma_f32_16x16x32_f16(false, af, false, bf, (short)0, c, false, false);
  }
  if (n < N){
    float bv = bias[n];
    #pragma unroll
    for (int r = 0; r < 8; ++r){
      int mr = mt*16 + r + (laneHi ? 8 : 0);
      P[mr*N + n] = c[r] + bv;
    }
  }
}

// ---------------- 4-qubit depth-2 circuit, one thread per sample ----------------
__global__ __launch_bounds__(256) void quantum_kernel(
    const float* __restrict__ P, float* __restrict__ evs)
{
  int b = blockIdx.x*256 + threadIdx.x;    // 1024 samples
  float2 st[16];
  #pragma unroll
  for (int i = 0; i < 16; ++i) st[i] = make_float2(0.f, 0.f);
  st[0].x = 1.f;
  #pragma unroll
  for (int d = 0; d < 2; ++d){
    #pragma unroll
    for (int q = 0; q < 4; ++q){
      const float* pp = P + ((b*2 + d)*4 + q)*3;
      float hx = 0.5f*pp[0], hy = 0.5f*pp[1], hz = 0.5f*pp[2];
      float cx = cosf(hx), sx = sinf(hx);
      float cy = cosf(hy), sy = sinf(hy);
      float cz = cosf(hz), sz = sinf(hz);
      float2 ez  = make_float2(cz, -sz);      // exp(-i hz)
      float2 cez = make_float2(cz,  sz);
      // m = RY @ RX
      float2 m00 = make_float2( cy*cx,  sy*sx);
      float2 m01 = make_float2(-sy*cx, -cy*sx);
      float2 m10 = make_float2( sy*cx, -cy*sx);
      float2 m11 = make_float2( cy*cx, -sy*sx);
      float2 U00 = cmul(ez,  m00), U01 = cmul(ez,  m01);
      float2 U10 = cmul(cez, m10), U11 = cmul(cez, m11);
      int stride = 8 >> q;                    // qubit q stride in flattened state
      #pragma unroll
      for (int i = 0; i < 16; ++i){
        if ((i & stride) == 0){
          float2 s0 = st[i], s1 = st[i + stride];
          st[i]          = cadd(cmul(U00, s0), cmul(U01, s1));
          st[i + stride] = cadd(cmul(U10, s0), cmul(U11, s1));
        }
      }
    }
    #pragma unroll
    for (int q = 0; q < 4; ++q){              // CNOT ring: control q, target (q+1)%4
      int t  = (q + 1) & 3;
      int cm = 8 >> q, tm = 8 >> t;
      #pragma unroll
      for (int i = 0; i < 16; ++i){
        if ((i & cm) && !(i & tm)){
          float2 tmp = st[i]; st[i] = st[i | tm]; st[i | tm] = tmp;
        }
      }
    }
  }
  #pragma unroll
  for (int q = 0; q < 4; ++q){
    float ev = 0.f;
    #pragma unroll
    for (int i = 0; i < 16; ++i){
      float pr = st[i].x*st[i].x + st[i].y*st[i].y;
      ev += (i & (8 >> q)) ? -pr : pr;
    }
    evs[b*4 + q] = ev;
  }
}

// ---------------- logits = evs @ Wc.T + bc, fused batch stats ----------------
__global__ __launch_bounds__(256) void logits_kernel(
    const float* __restrict__ evs, const float* __restrict__ Wc,
    const float* __restrict__ bc, float* __restrict__ logits,
    float* __restrict__ sums)
{
  __shared__ float ls[16];
  int tid = threadIdx.x;
  if (tid < 16) ls[tid] = 0.f;
  __syncthreads();
  int idx = blockIdx.x*256 + tid;    // b*8 + j
  int j = idx & 7;
  int b = idx >> 3;
  float v = bc[j];
  #pragma unroll
  for (int q = 0; q < 4; ++q) v += evs[b*4 + q]*Wc[j*4 + q];
  logits[idx] = v;
  atomicAdd(&ls[j], v);
  atomicAdd(&ls[8 + j], v*v);
  __syncthreads();
  if (tid < 16) atomicAdd(&sums[tid], ls[tid]);
}

__global__ __launch_bounds__(256) void bn_apply_kernel(
    const float* __restrict__ logits, const float* __restrict__ ss,
    float* __restrict__ out)
{
  int idx = blockIdx.x*256 + threadIdx.x;
  int j = idx & 7;
  out[idx] = logits[idx]*ss[j] + ss[8 + j];
}

__global__ __launch_bounds__(256) void f32_to_f16_kernel(
    const float* __restrict__ s, _Float16* __restrict__ d, int n)
{
  int i = blockIdx.x*256 + threadIdx.x;
  if (i < n) d[i] = (_Float16)s[i];
}

extern "C" void kernel_launch(void* const* d_in, const int* in_sizes, int n_in,
                              void* d_out, int out_size, void* d_ws, size_t ws_size,
                              hipStream_t stream)
{
  const float* x   = (const float*)d_in[0];
  const float* w1  = (const float*)d_in[1];
  const float* b1  = (const float*)d_in[2];
  const float* g1  = (const float*)d_in[3];
  const float* bt1 = (const float*)d_in[4];
  const float* w2  = (const float*)d_in[5];
  const float* b2  = (const float*)d_in[6];
  const float* g2  = (const float*)d_in[7];
  const float* bt2 = (const float*)d_in[8];
  const float* w3  = (const float*)d_in[9];
  const float* b3  = (const float*)d_in[10];
  const float* g3  = (const float*)d_in[11];
  const float* bt3 = (const float*)d_in[12];
  const float* We  = (const float*)d_in[13];
  const float* be  = (const float*)d_in[14];
  const float* Wc  = (const float*)d_in[15];
  const float* bc  = (const float*)d_in[16];
  const float* gn  = (const float*)d_in[17];
  const float* bnb = (const float*)d_in[18];
  float* out = (float*)d_out;

  char* ws = (char*)d_ws;
  // stats region (sums first so one memset clears all accumulators)
  float* sums1 = (float*)ws;             // 32
  float* sums2 = sums1 + 32;             // 64
  float* sums3 = sums2 + 64;             // 128
  float* sumsF = sums3 + 128;            // 16   (total 240 floats of accumulators)
  float* ss1   = sumsF + 16;             // 32
  float* ss2   = ss1 + 32;               // 64
  float* ss3   = ss2 + 64;               // 128
  float* ssF   = ss3 + 128;              // 16
  size_t off = 2048;
  float*    y1  = (float*)(ws + off);     off += (size_t)1024*16*28*28*4;
  _Float16* a1  = (_Float16*)(ws + off);  off += (size_t)1024*16*14*14*2;
  float*    y2  = (float*)(ws + off);     off += (size_t)1024*32*14*14*4;
  _Float16* a2  = (_Float16*)(ws + off);  off += (size_t)1024*32*7*7*2;
  float*    y3  = (float*)(ws + off);     off += (size_t)1024*64*7*7*4;
  _Float16* a3  = (_Float16*)(ws + off);  off += (size_t)1024*64*3*3*2;
  _Float16* w2h = (_Float16*)(ws + off);  off += (size_t)32*144*2;
  _Float16* w3h = (_Float16*)(ws + off);  off += (size_t)64*288*2;
  _Float16* Weh = (_Float16*)(ws + off);  off += (size_t)24*576*2;
  float*    P   = (float*)(ws + off);     off += (size_t)1024*24*4;
  float*    evs = (float*)(ws + off);     off += (size_t)1024*4*4;
  float*    lg  = (float*)(ws + off);     off += (size_t)1024*8*4;

  hipMemsetAsync(ws, 0, 240*sizeof(float), stream);   // zero all stat accumulators

  f32_to_f16_kernel<<<(4608  + 255)/256, 256, 0, stream>>>(w2, w2h, 4608);
  f32_to_f16_kernel<<<(18432 + 255)/256, 256, 0, stream>>>(w3, w3h, 18432);
  f32_to_f16_kernel<<<(13824 + 255)/256, 256, 0, stream>>>(We, Weh, 13824);

  // stage 1: 1->16 @ 28x28
  conv1_kernel<<<50176, 256, 0, stream>>>(x, w1, b1, y1, sums1);
  bn_finalize_kernel<<<1, 256, 0, stream>>>(sums1, g1, bt1, ss1, 16, 1.f/(1024.f*784.f));
  bn_relu_pool_kernel<16,28,28,14,14><<<12544, 256, 0, stream>>>(y1, ss1, a1);

  // stage 2: 16->32 @ 14x14 (WMMA: M=200704 N=32 K=144 -> 25088 waves)
  conv_wmma_kernel<16,14,14,32><<<3136, 256, 0, stream>>>(a1, w2h, b2, y2, sums2);
  bn_finalize_kernel<<<1, 256, 0, stream>>>(sums2, g2, bt2, ss2, 32, 1.f/(1024.f*196.f));
  bn_relu_pool_kernel<32,14,14,7,7><<<6272, 256, 0, stream>>>(y2, ss2, a2);

  // stage 3: 32->64 @ 7x7 (WMMA: M=50176 N=64 K=288 -> 12544 waves)
  conv_wmma_kernel<32,7,7,64><<<1568, 256, 0, stream>>>(a2, w3h, b3, y3, sums3);
  bn_finalize_kernel<<<1, 256, 0, stream>>>(sums3, g3, bt3, ss3, 64, 1.f/(1024.f*49.f));
  bn_relu_pool_kernel<64,7,7,3,3><<<2304, 256, 0, stream>>>(y3, ss3, a3);

  // head
  linear_wmma_kernel<<<16, 256, 0, stream>>>(a3, Weh, be, P);
  quantum_kernel<<<4, 256, 0, stream>>>(P, evs);
  logits_kernel<<<32, 256, 0, stream>>>(evs, Wc, bc, lg, sumsF);
  bn_finalize_kernel<<<1, 256, 0, stream>>>(sumsF, gn, bnb, ssF, 8, 1.f/1024.f);
  bn_apply_kernel<<<32, 256, 0, stream>>>(lg, ssF, out);

  (void)in_sizes; (void)n_in; (void)out_size; (void)ws_size;
}
